// RetrievalModule_74431783240449
// MI455X (gfx1250) — matrix operand order repaired and checked by reading, not compiled, not run on hardware
//
#include <hip/hip_runtime.h>
#include <hip/hip_bf16.h>

// ---------------- constants ----------------
#define NQ    256     // number of queries (fixed by problem)
#define DK    768     // embedding dim
#define KST   32      // K per WMMA step (bf16 16x16x32)
#define MT    64      // db rows per chunk (C tile: 256 x 64)
#define CHK   8       // chunks per block -> 512 db rows per block
#define T1    512     // threads in main kernel (16 waves)
#define BSTR  40      // Btile row stride in ushorts (pad vs bank conflicts)
#define SSTR  65      // scores row stride in floats (pad vs bank conflicts)

typedef __attribute__((ext_vector_type(16))) __bf16 v16bf;
typedef __attribute__((ext_vector_type(8)))  float  v8f;

union Frag { v16bf v; uint4 u[2]; };

// hardware RNE convert (v_cvt_pk_bf16_f32)
__device__ __forceinline__ unsigned int pack2bf(float a, float b) {
  union { __bf16 h[2]; unsigned int u; } x;
  x.h[0] = (__bf16)a;
  x.h[1] = (__bf16)b;
  return x.u;
}
__device__ __forceinline__ unsigned short f2bf(float a) {
  union { __bf16 h; unsigned short u; } x;
  x.h = (__bf16)a;
  return x.u;
}

// ---------------- kernel 0: query -> bf16 + 1/max(||q||,eps) ----------------
__global__ __launch_bounds__(256)
void retr_prep(const float* __restrict__ q, unsigned short* __restrict__ q16,
               float* __restrict__ qinv) {
  __shared__ float red[256];
  const int r = blockIdx.x, tid = threadIdx.x;
  float ss = 0.f;
  for (int k = tid; k < DK; k += 256) {
    float x = q[(long)r * DK + k];
    ss += x * x;
    q16[(long)r * DK + k] = f2bf(x);
  }
  red[tid] = ss;
  __syncthreads();
  for (int s = 128; s > 0; s >>= 1) {
    if (tid < s) red[tid] += red[tid + s];
    __syncthreads();
  }
  if (tid == 0) qinv[r] = 1.0f / fmaxf(sqrtf(red[0]), 1e-12f);
}

// ---------------- kernel 1: fused GEMM (bf16 WMMA) + norms + block top-16 ----------------
__global__ __launch_bounds__(T1, 1)
void retr_gemm_topk(const float* __restrict__ db,
                    const unsigned short* __restrict__ q16,
                    const float* __restrict__ qinv_g,
                    float* __restrict__ tks_out, int* __restrict__ tki_out,
                    int M) {
  __shared__ float tks[NQ * 16];
  __shared__ int   tki[NQ * 16];
  __shared__ float qinvS[NQ];
  __shared__ float rinv[MT];
  __shared__ __align__(16) unsigned char phase[NQ * SSTR * 4];  // scores, aliased over Btile
  unsigned short* Btile  = (unsigned short*)phase;              // [MT][BSTR] bf16
  float*          scores = (float*)phase;                       // [NQ][SSTR]

  const int tid  = threadIdx.x;
  const int wave = tid >> 5;
  const int lane = tid & 31;

  for (int i = tid; i < NQ * 16; i += T1) { tks[i] = -3.0e38f; tki[i] = -1; }
  if (tid < NQ) qinvS[tid] = qinv_g[tid];

  const long row_base = (long)blockIdx.x * (MT * CHK);

  // B staging map: 8 threads per db row, 4 floats each
  const int brow = tid >> 3;
  const int bc   = (tid & 7) * 4;
  // WMMA fragment lane mapping (per ISA 16-bit layouts)
  const int mA  = lane & 15;
  const int kbA = (lane >> 4) * 8;    // A: K base 0 or 8 (upper half at +16)
  const int nB  = lane & 15;
  const int kbB = (lane >> 4) * 8;    // B: K base 0 or 8 (upper half at +16)
  const unsigned short* aptr = q16 + (long)(wave * 16 + mA) * DK + kbA;

  __syncthreads();

  for (int c = 0; c < CHK; ++c) {
    const long row0 = row_base + (long)c * MT;
    v8f acc[4];
#pragma unroll
    for (int d = 0; d < 4; ++d)
#pragma unroll
      for (int v = 0; v < 8; ++v) acc[d][v] = 0.f;

    float rowsq = 0.f;
    const bool rv = (row0 + brow) < (long)M;
    const float* dptr = db + (row0 + brow) * (long)DK + bc;

#pragma clang loop unroll(disable)
    for (int kt = 0; kt < DK / KST; ++kt) {
      __syncthreads();
      // ---- stage B slice (fp32 -> bf16, hw cvt) + accumulate row sum-of-squares ----
      float4 f;
      if (rv) {
        f = *(const float4*)(dptr + kt * KST);
        __builtin_prefetch(dptr + (kt + 4) * KST, 0, 1);
      } else {
        f.x = f.y = f.z = f.w = 0.f;
      }
      rowsq += f.x * f.x + f.y * f.y + f.z * f.z + f.w * f.w;
      *(unsigned int*)(Btile + brow * BSTR + bc)     = pack2bf(f.x, f.y);
      *(unsigned int*)(Btile + brow * BSTR + bc + 2) = pack2bf(f.z, f.w);
      __syncthreads();

      // ---- fragments: A direct from global bf16 (L2-resident), B from LDS ----
      Frag af, bf[4];
      const unsigned short* ak = aptr + kt * KST;
      af.u[0] = *(const uint4*)(ak);        // K = kbA .. kbA+7
      af.u[1] = *(const uint4*)(ak + 16);   // K = kbA+16 .. kbA+23
#pragma unroll
      for (int d = 0; d < 4; ++d) {
        const unsigned short* bk = Btile + (d * 16 + nB) * BSTR + kbB;
        bf[d].u[0] = *(const uint4*)(bk);
        bf[d].u[1] = *(const uint4*)(bk + 16);
      }
#pragma unroll
      for (int d = 0; d < 4; ++d)
        acc[d] = __builtin_amdgcn_wmma_f32_16x16x32_bf16(
            false, af.v, false, bf[d].v, (short)0, acc[d], false, false);
    }
    __syncthreads();

    // ---- db row norms: reduce sum-of-squares across the 8 staging lanes ----
#pragma unroll
    for (int o = 1; o < 8; o <<= 1) rowsq += __shfl_xor(rowsq, o, 32);
    if ((tid & 7) == 0) rinv[brow] = rowsq;
    __syncthreads();
    if (tid < MT) rinv[tid] = 1.0f / fmaxf(sqrtf(rinv[tid]), 1e-12f);
    __syncthreads();

    // ---- scale + dump scores (C layout: VGPR v -> M = v + 8*(lane>=16), N = lane&15) ----
    {
      const int mb = (lane >> 4) * 8;
#pragma unroll
      for (int d = 0; d < 4; ++d) {
        const int n  = d * 16 + nB;
        const float rn = rinv[n];
#pragma unroll
        for (int v = 0; v < 8; ++v) {
          const int qq = wave * 16 + mb + v;
          scores[qq * SSTR + n] = acc[d][v] * qinvS[qq] * rn;
        }
      }
    }
    __syncthreads();

    // ---- per-query persistent top-16 over this chunk's 64 candidates ----
    if (tid < NQ) {
      float* L  = tks + tid * 16;
      int*   LI = tki + tid * 16;
      float thr = L[15];
      for (int n = 0; n < MT; ++n) {
        const long g = row0 + n;
        if (g >= (long)M) break;
        const float s = scores[tid * SSTR + n];
        if (s > thr) {
          L[15] = s; LI[15] = (int)g;
          for (int j = 15; j > 0; --j) {
            if (L[j] > L[j - 1]) {
              float t = L[j]; L[j] = L[j - 1]; L[j - 1] = t;
              int  ti = LI[j]; LI[j] = LI[j - 1]; LI[j - 1] = ti;
            } else break;
          }
          thr = L[15];
        }
      }
    }
    __syncthreads();
  }

  if (tid < NQ) {
    const long base = ((long)blockIdx.x * NQ + tid) * 16;
    for (int j = 0; j < 16; ++j) {
      tks_out[base + j] = tks[tid * 16 + j];
      tki_out[base + j] = tki[tid * 16 + j];
    }
  }
}

// ---------------- kernel 2: merge per-block candidates -> final top-16 ----------------
__global__ __launch_bounds__(256)
void retr_merge(const float* __restrict__ tks_ws, const int* __restrict__ tki_ws,
                float* __restrict__ out, int NB) {
  __shared__ float ls[256 * 16];
  __shared__ int   li[256 * 16];
  const int q = blockIdx.x, tid = threadIdx.x;

  float rs[16]; int ri[16];
#pragma unroll
  for (int j = 0; j < 16; ++j) { rs[j] = -3.0e38f; ri[j] = -1; }

  const int total = NB * 16;
  for (int c = tid; c < total; c += 256) {
    const int blk = c >> 4, j = c & 15;
    const long off = ((long)blk * NQ + q) * 16 + j;
    const float s = tks_ws[off];
    if (s > rs[15]) {
      rs[15] = s; ri[15] = tki_ws[off];
#pragma unroll
      for (int j2 = 15; j2 > 0; --j2) {
        if (rs[j2] > rs[j2 - 1]) {
          float t = rs[j2]; rs[j2] = rs[j2 - 1]; rs[j2 - 1] = t;
          int  ti = ri[j2]; ri[j2] = ri[j2 - 1]; ri[j2 - 1] = ti;
        }
      }
    }
  }
#pragma unroll
  for (int j = 0; j < 16; ++j) { ls[tid * 16 + j] = rs[j]; li[tid * 16 + j] = ri[j]; }
  __syncthreads();

  for (int st = 128; st >= 1; st >>= 1) {
    if (tid < st) {
      float* Ld = ls + tid * 16;        int* Id  = li + tid * 16;
      const float* Sx = ls + (tid + st) * 16;
      const int*  Sxi = li + (tid + st) * 16;
      for (int j = 0; j < 16; ++j) {
        const float s = Sx[j];
        if (s <= Ld[15]) break;          // src sorted desc -> safe early exit
        Ld[15] = s; Id[15] = Sxi[j];
        for (int jj = 15; jj > 0; --jj) {
          if (Ld[jj] > Ld[jj - 1]) {
            float t = Ld[jj]; Ld[jj] = Ld[jj - 1]; Ld[jj - 1] = t;
            int  ti = Id[jj]; Id[jj] = Id[jj - 1]; Id[jj - 1] = ti;
          } else break;
        }
      }
    }
    __syncthreads();
  }

  if (tid == 0) {
    for (int j = 0; j < 16; ++j) {
      out[q * 16 + j]            = ls[j];
      out[NQ * 16 + q * 16 + j]  = (float)li[j];   // indices half, float-cast (lossless < 2^24)
    }
  }
}

// ---------------- launch ----------------
extern "C" void kernel_launch(void* const* d_in, const int* in_sizes, int n_in,
                              void* d_out, int out_size, void* d_ws, size_t ws_size,
                              hipStream_t stream) {
  const float* query = (const float*)d_in[0];
  const float* db    = (const float*)d_in[1];
  const int N = in_sizes[0] / DK;   // 256
  const int M = in_sizes[1] / DK;   // 500000
  const int NB = (M + MT * CHK - 1) / (MT * CHK);

  char* w = (char*)d_ws;
  unsigned short* q16  = (unsigned short*)w;
  size_t off = (size_t)N * DK * sizeof(unsigned short);
  float* qinv = (float*)(w + off);
  off += (size_t)N * sizeof(float);
  off = (off + 255) & ~(size_t)255;
  float* tks_ws = (float*)(w + off);
  off += (size_t)NB * NQ * 16 * sizeof(float);
  int* tki_ws = (int*)(w + off);

  retr_prep<<<N, 256, 0, stream>>>(query, q16, qinv);
  retr_gemm_topk<<<NB, T1, 0, stream>>>(db, q16, qinv, tks_ws, tki_ws, M);
  retr_merge<<<NQ, 256, 0, stream>>>(tks_ws, tki_ws, (float*)d_out, NB);
}
